// ScaledDotProductAttention_22342419874385
// MI455X (gfx1250) — matrix operand order repaired and checked by reading, not compiled, not run on hardware
//
#include <hip/hip_runtime.h>

// tanh-attention for MI455X (gfx1250, wave32, WMMA).
// context = tanh(0.3/8 * Q K^T) @ V   — mask faithfully ignored (reference
// discards the masked_fill result).
//
// Compute-bound (AI ~515 FLOP/B): both GEMMs on v_wmma_f32_16x16x32_f16 with
// fp32 accumulation. Per wave: two 16-row M tiles -> 16 WMMAs per k-iteration.
// K/V staged via float4 register ping-pong (HBM latency hidden under WMMA),
// LDS rows padded for conflict-free b128 fragment gathers, tanh via
// v_exp_f32 + v_rcp_f32 (no IEEE-div chain).

typedef _Float16 f16;
typedef __attribute__((ext_vector_type(16))) _Float16 v16h;
typedef __attribute__((ext_vector_type(8)))  _Float16 h8;
typedef __attribute__((ext_vector_type(4)))  _Float16 h4;
typedef __attribute__((ext_vector_type(8)))  float    v8f;

namespace {

constexpr int kBatch  = 4;
constexpr int kHeads  = 16;
constexpr int kSeq    = 2048;
constexpr int kD      = 64;    // head dim
constexpr int kWaves  = 8;     // wave32 waves per workgroup
constexpr int kMT     = 2;     // 16-row M tiles per wave (32 query rows)
constexpr int kKT     = 32;    // keys per iteration (one WMMA K=32 chunk)
constexpr int kThreads = kWaves * 32;

// tanh(kScale*s) = 1 - 2/(exp2(kC*s)+1),  kC = 2*(0.3/8)*log2(e)
constexpr float kC = 0.075f * 1.44269504088896340736f;

// padded LDS row strides (halfs): bank-conflict-free gathers, 16B-aligned rows
constexpr int kKS  = 72;  // lds_k  rows (32 keys x 64 d)
constexpr int kVS  = 40;  // lds_vt rows (64 d x 32 keys)
constexpr int kPS  = 40;  // lds_p  rows (16 q x 32 keys) per wave

union Frag16 { v16h v; h8 h[2]; };

__device__ inline v8f zero8() {
  v8f z;
#pragma unroll
  for (int i = 0; i < 8; ++i) z[i] = 0.0f;
  return z;
}

__device__ inline v8f wmma_f16(const v16h& a, const v16h& b, const v8f& c) {
  return __builtin_amdgcn_wmma_f32_16x16x32_f16(false, a, false, b,
                                                (short)0, c, false, false);
}

__global__ __launch_bounds__(kThreads) void attn_tanh_wmma(
    const float* __restrict__ Q, const float* __restrict__ K,
    const float* __restrict__ V, float* __restrict__ Out)
{
  __shared__ __attribute__((aligned(16))) f16 lds_k [kKT * kKS];        // 4.5 KB
  __shared__ __attribute__((aligned(16))) f16 lds_vt[kD  * kVS];        // 5.0 KB
  __shared__ __attribute__((aligned(16))) f16 lds_p [kWaves * 16 * kPS];// 10 KB

  const int tid  = threadIdx.x;
  const int wave = tid >> 5;
  const int lane = tid & 31;
  const int lnm  = lane & 15;
  const int hi16 = lane >> 4;          // 0: lanes 0-15, 1: lanes 16-31

  const int head = blockIdx.y;                                  // b*H + h
  const int q0   = blockIdx.x * (kWaves * kMT * 16) + wave * (kMT * 16);

  const size_t headOff = (size_t)head * kSeq * kD;
  const float* Qh = Q + headOff;
  const float* Kh = K + headOff;
  const float* Vh = V + headOff;
  float*       Oh = Out + headOff;

  // ---- Persistent Q A-fragments built directly from global (f32 -> f16).
  // A(16x32,f16): lane m = lane&15; elems 0..7 -> d = 32c + (hi16?8:0)+e,
  // elems 8..15 -> d = 32c + 16 + (hi16?8:0) + (e-8): two 32B runs per lane.
  Frag16 aq[kMT][2];
#pragma unroll
  for (int mt = 0; mt < kMT; ++mt) {
#pragma unroll
    for (int c = 0; c < 2; ++c) {
      const float* qrow =
          Qh + (size_t)(q0 + mt * 16 + lnm) * kD + 32 * c + (hi16 ? 8 : 0);
      const float4 a0 = *reinterpret_cast<const float4*>(qrow + 0);
      const float4 a1 = *reinterpret_cast<const float4*>(qrow + 4);
      const float4 a2 = *reinterpret_cast<const float4*>(qrow + 16);
      const float4 a3 = *reinterpret_cast<const float4*>(qrow + 20);
      v16h& f = aq[mt][c].v;
      f[0]=(f16)a0.x;  f[1]=(f16)a0.y;  f[2]=(f16)a0.z;  f[3]=(f16)a0.w;
      f[4]=(f16)a1.x;  f[5]=(f16)a1.y;  f[6]=(f16)a1.z;  f[7]=(f16)a1.w;
      f[8]=(f16)a2.x;  f[9]=(f16)a2.y;  f[10]=(f16)a2.z; f[11]=(f16)a2.w;
      f[12]=(f16)a3.x; f[13]=(f16)a3.y; f[14]=(f16)a3.z; f[15]=(f16)a3.w;
    }
  }

  v8f acc[kMT][4];
#pragma unroll
  for (int mt = 0; mt < kMT; ++mt)
#pragma unroll
    for (int j = 0; j < 4; ++j) acc[mt][j] = zero8();

  // ---- Preload tile 0 into registers (float4 = global_load_b128).
  float4 kr[2], vr[2];
  {
    const float4* k4 = reinterpret_cast<const float4*>(Kh);
    const float4* v4 = reinterpret_cast<const float4*>(Vh);
    kr[0] = k4[tid]; kr[1] = k4[tid + kThreads];
    vr[0] = v4[tid]; vr[1] = v4[tid + kThreads];
  }

  f16* pb = lds_p + wave * 16 * kPS;   // per-wave P scratch

  for (int kt = 0; kt < kSeq / kKT; ++kt) {
    // ---- Spill staged tile kt (registers) into LDS as fp16.
#pragma unroll
    for (int b = 0; b < 2; ++b) {
      const int flat = b * 1024 + tid * 4;
      const int r = flat >> 6;          // key index
      const int c = flat & 63;          // d index (multiple of 4)
      h4 kh;
      kh[0]=(f16)kr[b].x; kh[1]=(f16)kr[b].y; kh[2]=(f16)kr[b].z; kh[3]=(f16)kr[b].w;
      *reinterpret_cast<h4*>(lds_k + r * kKS + c) = kh;   // ds_store_b64
      lds_vt[(c + 0) * kVS + r] = (f16)vr[b].x;           // transposed V
      lds_vt[(c + 1) * kVS + r] = (f16)vr[b].y;
      lds_vt[(c + 2) * kVS + r] = (f16)vr[b].z;
      lds_vt[(c + 3) * kVS + r] = (f16)vr[b].w;
    }
    __syncthreads();

    // ---- Issue next tile's global loads now; latency hides under WMMA.
    if (kt + 1 < kSeq / kKT) {
      const float4* k4 =
          reinterpret_cast<const float4*>(Kh + (size_t)(kt + 1) * kKT * kD);
      const float4* v4 =
          reinterpret_cast<const float4*>(Vh + (size_t)(kt + 1) * kKT * kD);
      kr[0] = k4[tid]; kr[1] = k4[tid + kThreads];
      vr[0] = v4[tid]; vr[1] = v4[tid + kThreads];
    }

#pragma unroll
    for (int mt = 0; mt < kMT; ++mt) {
      // scores(16x32) = Q(16x64) @ K^T ; tanh ; P -> per-wave LDS
#pragma unroll
      for (int t = 0; t < 2; ++t) {
        // B(32x16): lane holds col n = 16t+lnm, d-rows (hi16?8:0)+0..7 and 16+(hi16?8:0)+0..7
        const f16* krow = lds_k + (t * 16 + lnm) * kKS + (hi16 ? 8 : 0);
        Frag16 bk0, bk1;
        bk0.h[0] = *reinterpret_cast<const h8*>(krow);
        bk0.h[1] = *reinterpret_cast<const h8*>(krow + 16);
        bk1.h[0] = *reinterpret_cast<const h8*>(krow + 32);
        bk1.h[1] = *reinterpret_cast<const h8*>(krow + 48);

        v8f s = zero8();
        s = wmma_f16(aq[mt][0].v, bk0.v, s);
        s = wmma_f16(aq[mt][1].v, bk1.v, s);
#pragma unroll
        for (int g = 0; g < 8; ++g) {
          // tanh(kScale*s) = 1 - 2*rcp(exp2(kC*s)+1): v_exp_f32 + v_rcp_f32
          const float t2 = __builtin_amdgcn_exp2f(kC * s[g]);
          const float p  = 1.0f - 2.0f * __builtin_amdgcn_rcpf(t2 + 1.0f);
          pb[(g + (hi16 ? 8 : 0)) * kPS + t * 16 + lnm] = (f16)p;
        }
      }

      // A fragment of P(16x32) via per-wave LDS round trip (in-order per wave)
      Frag16 ap;
      {
        const f16* pr = pb + lnm * kPS + (hi16 ? 8 : 0);
        ap.h[0] = *reinterpret_cast<const h8*>(pr);
        ap.h[1] = *reinterpret_cast<const h8*>(pr + 16);
      }

      // context(16x64) += P(16x32) @ V(32x64)
#pragma unroll
      for (int j = 0; j < 4; ++j) {
        const f16* vrow = lds_vt + (j * 16 + lnm) * kVS + (hi16 ? 8 : 0);
        Frag16 bv;
        bv.h[0] = *reinterpret_cast<const h8*>(vrow);
        bv.h[1] = *reinterpret_cast<const h8*>(vrow + 16);
        acc[mt][j] = wmma_f16(ap.v, bv.v, acc[mt][j]);
      }
    }
    __syncthreads();   // protect lds_k/lds_vt before next spill
  }

  // ---- Epilogue: 32x64 fp32 tile. C/D layout: vgpr g -> row g+8*hi16.
  float* obase = Oh + (size_t)q0 * kD;
#pragma unroll
  for (int mt = 0; mt < kMT; ++mt)
#pragma unroll
    for (int j = 0; j < 4; ++j)
#pragma unroll
      for (int g = 0; g < 8; ++g)
        obase[(mt * 16 + g + (hi16 ? 8 : 0)) * kD + j * 16 + lnm] =
            acc[mt][j][g];
}

} // namespace

extern "C" void kernel_launch(void* const* d_in, const int* in_sizes, int n_in,
                              void* d_out, int out_size, void* d_ws, size_t ws_size,
                              hipStream_t stream) {
  const float* Q = (const float*)d_in[0];
  const float* K = (const float*)d_in[1];
  const float* V = (const float*)d_in[2];
  // d_in[3] (attn_mask) intentionally unused: the reference's masked_fill
  // result is discarded, so the mask has no effect on the output.
  float* Out = (float*)d_out;
  (void)in_sizes; (void)n_in; (void)d_ws; (void)ws_size; (void)out_size;

  dim3 grid(kSeq / (kWaves * kMT * 16), kBatch * kHeads);  // (8, 64)
  dim3 block(kThreads);                                    // 256 = 8 wave32
  hipLaunchKernelGGL(attn_tanh_wmma, grid, block, 0, stream, Q, K, V, Out);
}
